// MambaLayer_17377437680463
// MI455X (gfx1250) — matrix operand order repaired and checked by reading, not compiled, not run on hardware
//
#include <hip/hip_runtime.h>
#include <hip/hip_bf16.h>
#include <math.h>

// ---------------- problem constants ----------------
#define BB      2
#define LL      2048
#define DD      512
#define DIN     1024          // 2*D
#define DST     16
#define DCONV   4
#define DTR     32            // ceil(D/16)
#define NT      (BB*LL)       // 4096 tokens

typedef __bf16 bf16_t;
typedef bf16_t v16bf __attribute__((ext_vector_type(16)));
typedef float  v8f   __attribute__((ext_vector_type(8)));
typedef unsigned uint32x4 __attribute__((ext_vector_type(4)));
typedef int      int32x8  __attribute__((ext_vector_type(8)));
typedef int      int32x4  __attribute__((ext_vector_type(4)));

#if __has_builtin(__builtin_amdgcn_tensor_load_to_lds)
#define HAVE_TDM 1
#else
#define HAVE_TDM 0
#endif

// ---------------------------------------------------------------------
// TDM 2D tile load: tile is (tile_rows x 32) bf16 elements, row-major,
// row stride = rowlen elements, into LDS at ldsaddr (contiguous rows).
// D# packing per CDNA5 ISA ch.8 (group0: count/lds/global/type=2,
// group1: data_size=2B, tensor dims, tile dims, dim0 stride).
// ---------------------------------------------------------------------
__device__ __forceinline__ void tdm_load_tile(const bf16_t* gptr, unsigned ldsaddr,
                                              int rowlen, int nrows_tensor, int tile_rows)
{
#if HAVE_TDM
    unsigned long long ga = (unsigned long long)(uintptr_t)gptr;
    unsigned td0 = (unsigned)rowlen;
    unsigned td1 = (unsigned)nrows_tensor;
    unsigned long long st0 = (unsigned long long)(unsigned)rowlen;

    uint32x4 g0;
    g0[0] = 1u;                                               // count=1
    g0[1] = ldsaddr;                                          // lds_addr
    g0[2] = (unsigned)ga;                                     // global_addr lo
    g0[3] = ((unsigned)(ga >> 32) & 0x01FFFFFFu) | (2u << 30);// addr hi | type=2

    int32x8 g1;
    g1[0] = (int)(1u << 16);                                  // data_size=1 (2B)
    g1[1] = (int)((td0 & 0xFFFFu) << 16);                     // tensor_dim0 lo16
    g1[2] = (int)(((td0 >> 16) & 0xFFFFu) | ((td1 & 0xFFFFu) << 16));
    g1[3] = (int)(((td1 >> 16) & 0xFFFFu) | (32u << 16));     // tile_dim0=32
    g1[4] = tile_rows;                                        // tile_dim1
    g1[5] = (int)(unsigned)(st0 & 0xFFFFFFFFu);               // dim0 stride lo32
    g1[6] = (int)(unsigned)((st0 >> 32) & 0xFFFFu);           // dim0 stride hi16
    g1[7] = 0;

    int32x4 z4 = {0, 0, 0, 0};
    int32x8 z8 = {0, 0, 0, 0, 0, 0, 0, 0};
    __builtin_amdgcn_tensor_load_to_lds(g0, g1, z4, z4, z8, 0);
#else
    (void)gptr; (void)ldsaddr; (void)rowlen; (void)nrows_tensor; (void)tile_rows;
#endif
}

// =====================================================================
// bf16 WMMA GEMM: C[M,N] = A[M,K] * Bt[N,K]^T (+Add).
// A row-major bf16 [M][K]; Bt pre-transposed bf16 [N][K] so both LDS
// tiles are row-major 2D TDM tiles. Double-buffered LDS, TDM prefetch
// of the next K tile overlapped with WMMA on the current one.
// Block: 256 thr = 8 waves (4 M x 2 N-groups); wave = 16 x 16*WN -> WN
// WMMAs per 32-deep K step with a shared A fragment.
// =====================================================================
template<int WN, bool WRITE_BF>
__global__ void __launch_bounds__(256)
gemm_wmma(const bf16_t* __restrict__ A, int lda,
          const bf16_t* __restrict__ Bt, int ldbt,
          float* __restrict__ C, int ldc,
          bf16_t* __restrict__ Cbf,
          int M, int N, int K,
          const float* __restrict__ Add, int ldadd)
{
    constexpr int BN = 32 * WN;
    __shared__ bf16_t Al[2][64][32];
    __shared__ bf16_t Bl[2][BN][32];

    const int tid  = threadIdx.x;
    const int lane = tid & 31;
    const int wid  = tid >> 5;
    const int wm   = wid & 3;            // wave M tile (0..3)
    const int wn   = wid >> 2;           // wave N group (0..1)
    const int gm   = blockIdx.y * 64;
    const int gn   = blockIdx.x * BN;

    const int lrow = lane & 15;
    const int half = lane >> 4;

    v8f acc[WN];
    #pragma unroll
    for (int j = 0; j < WN; ++j) acc[j] = (v8f){0.f,0.f,0.f,0.f,0.f,0.f,0.f,0.f};

    const int nsteps = K >> 5;

#if HAVE_TDM
    // prologue: fill buffer 0
    if (wid == 0)
        tdm_load_tile(A  + (size_t)gm * lda,  (unsigned)(uintptr_t)&Al[0][0][0], lda,  M, 64);
    if (wid == 1)
        tdm_load_tile(Bt + (size_t)gn * ldbt, (unsigned)(uintptr_t)&Bl[0][0][0], ldbt, N, BN);
#endif

    for (int i = 0; i < nsteps; ++i) {
        const int cur = i & 1;
        const int k0  = i << 5;

#if HAVE_TDM
        if (i + 1 < nsteps) {
            if (wid == 0)
                tdm_load_tile(A  + (size_t)gm * lda  + (k0 + 32),
                              (unsigned)(uintptr_t)&Al[cur ^ 1][0][0], lda,  M, 64);
            if (wid == 1)
                tdm_load_tile(Bt + (size_t)gn * ldbt + (k0 + 32),
                              (unsigned)(uintptr_t)&Bl[cur ^ 1][0][0], ldbt, N, BN);
            if (wid < 2) __builtin_amdgcn_s_wait_tensorcnt(1);
        } else {
            if (wid < 2) __builtin_amdgcn_s_wait_tensorcnt(0);
        }
#else
        // manual staging fallback: 16B vector copies (tiles are row-contiguous)
        {
            int c = tid, r = c >> 2, kq = (c & 3) << 3;      // A: 64 rows x 4 chunks
            *(uint4*)&Al[cur][r][kq] = *(const uint4*)&A[(size_t)(gm + r) * lda + k0 + kq];
        }
        #pragma unroll
        for (int j = 0; j < (BN * 4) / 256; ++j) {           // B: BN rows x 4 chunks
            int c = tid + 256 * j;
            int n = c >> 2, kq = (c & 3) << 3;
            *(uint4*)&Bl[cur][n][kq] = *(const uint4*)&Bt[(size_t)(gn + n) * ldbt + k0 + kq];
        }
#endif
        __syncthreads();

        // ---- A fragment (shared across the WN wmmas) ----
        union { v16bf v; unsigned u[8]; } af;
        const unsigned* ap = (const unsigned*)(&Al[cur][wm * 16 + lrow][0]);
        #pragma unroll
        for (int v = 0; v < 8; ++v) {
            int ak = ((v < 4) ? (2 * v) : (16 + 2 * (v - 4))) + 8 * half;
            af.u[v] = ap[ak >> 1];
        }

        #pragma unroll
        for (int j = 0; j < WN; ++j) {
            union { v16bf v; unsigned u[8]; } bfm;
            const unsigned* bp = (const unsigned*)(&Bl[cur][wn * WN * 16 + j * 16 + lrow][0]);
            #pragma unroll
            for (int v = 0; v < 8; ++v)
                bfm.u[v] = bp[((v < 4) ? v : (8 + (v - 4))) + 4 * half];
            acc[j] = __builtin_amdgcn_wmma_f32_16x16x32_bf16(
                         false, af.v, false, bfm.v, (short)0, acc[j], false, false);
        }

        if (i + 1 < nsteps) __syncthreads();   // release buffer for next TDM write
    }

    // ---- epilogue: VGPR r -> M = 8*half + r, N = lane&15 ----
    #pragma unroll
    for (int j = 0; j < WN; ++j) {
        const int ccol = gn + wn * WN * 16 + j * 16 + lrow;
        #pragma unroll
        for (int r = 0; r < 8; ++r) {
            int crow = gm + wm * 16 + half * 8 + r;
            float val = acc[j][r];
            if (Add) val += Add[(size_t)crow * ldadd + ccol];
            C[(size_t)crow * ldc + ccol] = val;
            if (WRITE_BF) Cbf[(size_t)crow * ldc + ccol] = (bf16_t)val;
        }
    }
}

// ---------------- fp32 -> bf16 elementwise ----------------
__global__ void __launch_bounds__(256)
cvt_bf16_kernel(const float* __restrict__ src, bf16_t* __restrict__ dst, int n)
{
    int i = blockIdx.x * 256 + threadIdx.x;
    if (i < n) dst[i] = (bf16_t)src[i];
}

// ---------------- fp32 [K][N] -> bf16 transposed [N][K] ----------------
__global__ void __launch_bounds__(256)
transpose_cvt_kernel(const float* __restrict__ src, bf16_t* __restrict__ dst,
                     int K, int N)
{
    __shared__ float tile[32][33];
    const int n0 = blockIdx.x * 32;
    const int k0 = blockIdx.y * 32;
    const int tx = threadIdx.x & 31;
    const int ty = threadIdx.x >> 5;          // 0..7
    #pragma unroll
    for (int i = 0; i < 32; i += 8)
        tile[ty + i][tx] = src[(size_t)(k0 + ty + i) * N + (n0 + tx)];
    __syncthreads();
    #pragma unroll
    for (int i = 0; i < 32; i += 8)
        dst[(size_t)(n0 + ty + i) * K + (k0 + tx)] = (bf16_t)tile[tx][ty + i];
}

// ---------------- depthwise causal conv1d(4) + SiLU ----------------
__global__ void __launch_bounds__(256)
conv_silu_kernel(const float* __restrict__ xz,
                 const float* __restrict__ conv_w,
                 const float* __restrict__ conv_b,
                 float* __restrict__ ucv,
                 bf16_t* __restrict__ ucv_bf)
{
    int idx = blockIdx.x * 256 + threadIdx.x;       // (b,l,d), d fastest
    int d = idx & (DIN - 1);
    int l = (idx >> 10) & (LL - 1);
    int b = idx >> 21;

    float acc = conv_b[d];
    #pragma unroll
    for (int j = 0; j < DCONV; ++j) {
        int ls = l - (DCONV - 1) + j;
        if (ls >= 0)
            acc += conv_w[d * DCONV + j] *
                   xz[((size_t)(b * LL + ls)) * (2 * DIN) + d];
    }
    float s = acc / (1.f + __expf(-acc));           // SiLU
    ucv[(size_t)idx]    = s;
    ucv_bf[(size_t)idx] = (bf16_t)s;
}

// ---------------- selective scan, fused gating ----------------
__global__ void __launch_bounds__(256)
scan_kernel(const float* __restrict__ dtp,
            const float* __restrict__ ucv,
            const float* __restrict__ xdbl,
            const float* __restrict__ xz,
            const float* __restrict__ A_log,
            const float* __restrict__ Dp,
            const float* __restrict__ bdt,
            bf16_t* __restrict__ y_bf)
{
    int t = blockIdx.x * 256 + threadIdx.x;  // 0..2047
    int b = t >> 10;
    int d = t & (DIN - 1);

    float Acoef[DST], h[DST];
    #pragma unroll
    for (int s = 0; s < DST; ++s) {
        Acoef[s] = -__expf(A_log[d * DST + s]);
        h[s] = 0.f;
    }
    const float bias = bdt[d];
    const float Dpar = Dp[d];

    for (int l = 0; l < LL; ++l) {
        size_t tok = (size_t)(b * LL + l);
        float dtv = dtp[tok * DIN + d] + bias;
        dtv = (dtv > 20.f) ? dtv : log1pf(__expf(dtv));   // softplus
        float uv = ucv[tok * DIN + d];
        float zv = xz[tok * (2 * DIN) + DIN + d];

        const float* Bt = &xdbl[tok * 64 + DTR];
        const float* Ct = Bt + DST;

        float du  = dtv * uv;
        float out = 0.f;
        #pragma unroll
        for (int s = 0; s < DST; ++s) {
            h[s] = h[s] * __expf(dtv * Acoef[s]) + du * Bt[s];
            out += h[s] * Ct[s];
        }
        float yv = out + uv * Dpar;
        float sz = zv / (1.f + __expf(-zv));              // silu(z)
        y_bf[tok * DIN + d] = (bf16_t)(yv * sz);
    }
}

// =====================================================================
// Launch pipeline
// =====================================================================
extern "C" void kernel_launch(void* const* d_in, const int* in_sizes, int n_in,
                              void* d_out, int out_size, void* d_ws, size_t ws_size,
                              hipStream_t stream)
{
    const float* x      = (const float*)d_in[0];  // (B,L,512)
    const float* W_in   = (const float*)d_in[1];  // (512,2048)
    const float* conv_w = (const float*)d_in[2];  // (1024,4)
    const float* conv_b = (const float*)d_in[3];  // (1024,)
    const float* W_x    = (const float*)d_in[4];  // (1024,64)
    const float* W_dt   = (const float*)d_in[5];  // (32,1024)
    const float* b_dt   = (const float*)d_in[6];  // (1024,)
    const float* A_log  = (const float*)d_in[7];  // (1024,16)
    const float* D_par  = (const float*)d_in[8];  // (1024,)
    const float* W_out  = (const float*)d_in[9];  // (1024,512)
    float* out = (float*)d_out;                   // (B,L,512)

    // ---- workspace layout ----
    float* ws   = (float*)d_ws;
    float* xz   = ws;                                  // NT x 2048 f32
    float* ucv  = xz   + (size_t)NT * (2 * DIN);       // NT x 1024 f32
    float* xdbl = ucv  + (size_t)NT * DIN;             // NT x 64   f32
    float* dtp  = xdbl + (size_t)NT * 64;              // NT x 1024 f32
    bf16_t* bfp  = (bf16_t*)(dtp + (size_t)NT * DIN);
    bf16_t* xbf     = bfp;                             // NT x 512
    bf16_t* W_inT   = xbf    + (size_t)NT * DD;        // 2048 x 512  (B^T)
    bf16_t* W_xT    = W_inT  + (size_t)DD * 2 * DIN;   // 64 x 1024
    bf16_t* W_dtT   = W_xT   + (size_t)DIN * 64;       // 1024 x 32
    bf16_t* W_outT  = W_dtT  + (size_t)DTR * DIN;      // 512 x 1024
    bf16_t* ucvbf   = W_outT + (size_t)DIN * DD;       // NT x 1024
    bf16_t* xdblbf  = ucvbf  + (size_t)NT * DIN;       // NT x 64
    bf16_t* ybbf    = xdblbf + (size_t)NT * 64;        // NT x 1024

    // ---- 0) one-time operand conversion: A-side bf16, B-side bf16 transposed ----
    cvt_bf16_kernel<<<(NT * DD + 255) / 256, 256, 0, stream>>>(x, xbf, NT * DD);
    transpose_cvt_kernel<<<dim3((2 * DIN) / 32, DD / 32), 256, 0, stream>>>(W_in, W_inT, DD, 2 * DIN);
    transpose_cvt_kernel<<<dim3(64 / 32, DIN / 32), 256, 0, stream>>>(W_x, W_xT, DIN, 64);
    transpose_cvt_kernel<<<dim3(DIN / 32, DTR / 32), 256, 0, stream>>>(W_dt, W_dtT, DTR, DIN);
    transpose_cvt_kernel<<<dim3(DD / 32, DIN / 32), 256, 0, stream>>>(W_out, W_outT, DIN, DD);

    // ---- 1) xz = x @ W_in        [4096x512]x[512x2048] ----
    gemm_wmma<4, false><<<dim3((2 * DIN) / 128, NT / 64), 256, 0, stream>>>(
        xbf, DD, W_inT, DD, xz, 2 * DIN, nullptr, NT, 2 * DIN, DD, nullptr, 0);

    // ---- 2) depthwise conv + SiLU -> ucv (f32 + bf16) ----
    conv_silu_kernel<<<(NT * DIN) / 256, 256, 0, stream>>>(xz, conv_w, conv_b, ucv, ucvbf);

    // ---- 3) x_dbl = ucv @ W_x    [4096x1024]x[1024x64], dual f32/bf16 out ----
    gemm_wmma<2, true><<<dim3(64 / 64, NT / 64), 256, 0, stream>>>(
        ucvbf, DIN, W_xT, DIN, xdbl, 64, xdblbf, NT, 64, DIN, nullptr, 0);

    // ---- 4) dt_pre = dt_r @ W_dt [4096x32]x[32x1024] (dt_r = xdbl[:, :32]) ----
    gemm_wmma<4, false><<<dim3(DIN / 128, NT / 64), 256, 0, stream>>>(
        xdblbf, 64, W_dtT, DTR, dtp, DIN, nullptr, NT, DIN, DTR, nullptr, 0);

    // ---- 5) selective scan + gating -> yb (bf16) ----
    scan_kernel<<<(BB * DIN) / 256, 256, 0, stream>>>(
        dtp, ucv, xdbl, xz, A_log, D_par, b_dt, ybbf);

    // ---- 6) out = x + yb @ W_out [4096x1024]x[1024x512], fused residual ----
    gemm_wmma<4, false><<<dim3(DD / 128, NT / 64), 256, 0, stream>>>(
        ybbf, DIN, W_outT, DIN, out, DD, nullptr, NT, DD, DIN, x, DD);

    (void)in_sizes; (void)n_in; (void)out_size; (void)ws_size;
}
